// GCNEncoder_87239375716439
// MI455X (gfx1250) — matrix operand order repaired and checked by reading, not compiled, not run on hardware
//
#include <hip/hip_runtime.h>
#include <hip/hip_bf16.h>

// ---------------------------------------------------------------------------
// 2-layer GCN for MI455X (gfx1250, wave32).
//  - GEMMs via v_wmma_f32_16x16x32_bf16 with bf16 hi/lo split (~fp32 accuracy)
//  - edge gather/scatter via float4 loads + native f32 global atomics
// ---------------------------------------------------------------------------

typedef __attribute__((ext_vector_type(16))) __bf16 v16bf;
typedef __attribute__((ext_vector_type(8)))  __bf16 v8bf;
typedef __attribute__((ext_vector_type(8)))  float  v8f;

static constexpr int KDIM = 128;     // both layers have K = 128

// ------------------------------- degree ------------------------------------

__global__ void deg_init_kernel(float* __restrict__ deg, int N) {
  int i = blockIdx.x * blockDim.x + threadIdx.x;
  if (i < N) deg[i] = 1.0f;                      // self-loop contribution
}

__global__ void deg_scatter_kernel(const long long* __restrict__ row,
                                   float* __restrict__ deg, int E) {
  int e = blockIdx.x * blockDim.x + threadIdx.x;
  if (e < E) unsafeAtomicAdd(&deg[(int)row[e]], 1.0f);
}

__global__ void deg_finalize_kernel(float* __restrict__ deg, int N) {
  int i = blockIdx.x * blockDim.x + threadIdx.x;
  if (i < N) deg[i] = rsqrtf(deg[i]);            // deg >= 1 always (self-loops)
}

// ------------------------------- GEMM --------------------------------------
// H[N, COLS] = X[N, 128] @ W[128, COLS], fp32 in/out, bf16x3 WMMA inside.
// One wave computes one 16x16 output tile; a block of (2*COLS) threads covers
// all COLS for a 16-row tile and grid-strides over row tiles.

template <int COLS, bool RELU_IN>
__global__ __launch_bounds__(2 * COLS)
void gcn_gemm_kernel(const float* __restrict__ X,
                     const float* __restrict__ W,
                     float* __restrict__ H,
                     int N, int nTiles) {
  constexpr int LDK = KDIM + 8;                  // 136 bf16 = 272 B, 16B-aligned rows

  __shared__ __attribute__((aligned(16))) __bf16 sW_hi[COLS][LDK];  // transposed: [n][k]
  __shared__ __attribute__((aligned(16))) __bf16 sW_lo[COLS][LDK];
  __shared__ __attribute__((aligned(16))) __bf16 sX_hi[16][LDK];    // [m][k]
  __shared__ __attribute__((aligned(16))) __bf16 sX_lo[16][LDK];

  const int tid      = threadIdx.x;
  const int nThreads = 2 * COLS;
  const int wave     = tid >> 5;
  const int lane     = tid & 31;
  const int half     = lane >> 4;                // lane 0-15 vs 16-31
  const int lidx     = lane & 15;

  // Stage W transposed into LDS (hi/lo bf16 split). W is [k][n] row-major.
  for (int idx = tid; idx < KDIM * COLS; idx += nThreads) {
    const int k = idx / COLS, n = idx % COLS;
    const float w  = W[idx];
    const __bf16 h = (__bf16)w;
    sW_hi[n][k] = h;
    sW_lo[n][k] = (__bf16)(w - (float)h);
  }
  __syncthreads();

  for (int tile = blockIdx.x; tile < nTiles; tile += gridDim.x) {
    const int rowBase = tile * 16;

    // Stage 16x128 tile of X (fused ReLU for layer-2 input), hi/lo split.
    for (int idx = tid; idx < 16 * KDIM; idx += nThreads) {
      const int r = idx >> 7, k = idx & (KDIM - 1);
      int gr = rowBase + r; if (gr >= N) gr = N - 1;     // clamp (store is guarded)
      float v = X[(size_t)gr * KDIM + k];
      if (RELU_IN) v = fmaxf(v, 0.0f);
      const __bf16 h = (__bf16)v;
      sX_hi[r][k] = h;
      sX_lo[r][k] = (__bf16)(v - (float)h);
    }
    // Prefetch next tile's rows while we compute (global_prefetch_b8).
    {
      const int nextTile = tile + gridDim.x;
      if (nextTile < nTiles && tid < 16)
        __builtin_prefetch(&X[(size_t)(nextTile * 16 + tid) * KDIM], 0, 1);
    }
    __syncthreads();

    v8f acc = {};                                // 16x16 f32 accumulator tile
    const int n = wave * 16 + lidx;              // this lane's output column

#pragma unroll
    for (int kt = 0; kt < 4; ++kt) {
      // A fragment (16x32 bf16): elements 0..7 -> K = kt*32 + 8*half + e,
      //                          elements 8..15 -> +16 (ISA 16-bit A layout).
      const int ka = kt * 32 + 8 * half;
      const v8bf a0h = *(const v8bf*)&sX_hi[lidx][ka];
      const v8bf a1h = *(const v8bf*)&sX_hi[lidx][ka + 16];
      const v8bf a0l = *(const v8bf*)&sX_lo[lidx][ka];
      const v8bf a1l = *(const v8bf*)&sX_lo[lidx][ka + 16];
      const v16bf a_hi = __builtin_shufflevector(a0h, a1h, 0,1,2,3,4,5,6,7,8,9,10,11,12,13,14,15);
      const v16bf a_lo = __builtin_shufflevector(a0l, a1l, 0,1,2,3,4,5,6,7,8,9,10,11,12,13,14,15);

      // B fragment (32x16 bf16): elements 0..7 -> K = kt*32 + 8*half + e,
      //                          elements 8..15 -> +16 (same as A).
      const int kb = kt * 32 + 8 * half;
      const v8bf b0h = *(const v8bf*)&sW_hi[n][kb];
      const v8bf b1h = *(const v8bf*)&sW_hi[n][kb + 16];
      const v8bf b0l = *(const v8bf*)&sW_lo[n][kb];
      const v8bf b1l = *(const v8bf*)&sW_lo[n][kb + 16];
      const v16bf b_hi = __builtin_shufflevector(b0h, b1h, 0,1,2,3,4,5,6,7,8,9,10,11,12,13,14,15);
      const v16bf b_lo = __builtin_shufflevector(b0l, b1l, 0,1,2,3,4,5,6,7,8,9,10,11,12,13,14,15);

      // bf16x3: x*w ~= hi*hi + hi*lo + lo*hi  (f32 accumulate)
      acc = __builtin_amdgcn_wmma_f32_16x16x32_bf16(false, a_hi, false, b_hi, (short)0, acc, false, false);
      acc = __builtin_amdgcn_wmma_f32_16x16x32_bf16(false, a_hi, false, b_lo, (short)0, acc, false, false);
      acc = __builtin_amdgcn_wmma_f32_16x16x32_bf16(false, a_lo, false, b_hi, (short)0, acc, false, false);
    }

    // C/D layout: VGPR r -> M = r + 8*half, N = lane%16.
#pragma unroll
    for (int r = 0; r < 8; ++r) {
      const int m = rowBase + r + 8 * half;
      if (m < N) H[(size_t)m * COLS + n] = acc[r];
    }
    __syncthreads();                             // protect sX before next tile
  }
}

// --------------------------- aggregation -----------------------------------
// out[i] = h[i] * dinv[i]^2 + bias   (self-loop message + bias), float4-wide.

template <int C>
__global__ void agg_init_kernel(const float* __restrict__ h,
                                const float* __restrict__ dinv,
                                const float* __restrict__ bias,
                                float* __restrict__ out, int N) {
  constexpr int G = C / 4;
  int idx = blockIdx.x * blockDim.x + threadIdx.x;   // over N*C/4
  if (idx >= N * G) return;
  const int i  = idx / G;
  const int c4 = idx % G;
  const float d = dinv[i];
  const float s = d * d;
  const float4 hv = ((const float4*)h)[idx];
  const float4 bv = ((const float4*)bias)[c4];
  float4 o;
  o.x = hv.x * s + bv.x;
  o.y = hv.y * s + bv.y;
  o.z = hv.z * s + bv.z;
  o.w = hv.w * s + bv.w;
  ((float4*)out)[idx] = o;
}

// out[row[e]] += h[col[e]] * dinv[row]*dinv[col]; C/4 threads cooperate per edge.
template <int C>
__global__ void agg_edge_kernel(const float* __restrict__ h,
                                const long long* __restrict__ row,
                                const long long* __restrict__ col,
                                const float* __restrict__ dinv,
                                float* __restrict__ out, int E) {
  constexpr int G = C / 4;
  const long long tid = (long long)blockIdx.x * blockDim.x + threadIdx.x;
  const int e = (int)(tid / G);
  const int g = (int)(tid % G);
  if (e >= E) return;
  const int r = (int)row[e];
  const int c = (int)col[e];
  const float nrm = dinv[r] * dinv[c];
  const float4 v = ((const float4*)(h + (size_t)c * C))[g];
  float* o = out + (size_t)r * C + g * 4;
  unsafeAtomicAdd(o + 0, v.x * nrm);
  unsafeAtomicAdd(o + 1, v.y * nrm);
  unsafeAtomicAdd(o + 2, v.z * nrm);
  unsafeAtomicAdd(o + 3, v.w * nrm);
}

// ------------------------------- launch ------------------------------------

extern "C" void kernel_launch(void* const* d_in, const int* in_sizes, int n_in,
                              void* d_out, int out_size, void* d_ws, size_t ws_size,
                              hipStream_t stream) {
  const float*     x   = (const float*)d_in[0];
  const long long* ei  = (const long long*)d_in[1];   // int64 [2, E]
  const float*     W1  = (const float*)d_in[2];
  const float*     b1  = (const float*)d_in[3];
  const float*     W2  = (const float*)d_in[4];
  const float*     b2  = (const float*)d_in[5];
  float*           out = (float*)d_out;

  const int N = in_sizes[0] / 128;     // 50000
  const int E = in_sizes[1] / 2;       // 800000
  const long long* rowp = ei;          // targets
  const long long* colp = ei + E;      // sources

  // Workspace layout (256B aligned).
  auto align_up = [](size_t v) { return (v + 255) & ~(size_t)255; };
  char* ws = (char*)d_ws;
  size_t off = 0;
  float* dinv = (float*)(ws + off); off += align_up((size_t)N * sizeof(float));
  float* h1   = (float*)(ws + off); off += align_up((size_t)N * 128 * sizeof(float));
  float* agg1 = (float*)(ws + off); off += align_up((size_t)N * 128 * sizeof(float));
  float* h2   = (float*)(ws + off);

  const int nTiles = (N + 15) / 16;

  // 1) dinv = rsqrt(deg) with self-loops
  deg_init_kernel<<<(N + 255) / 256, 256, 0, stream>>>(dinv, N);
  deg_scatter_kernel<<<(E + 255) / 256, 256, 0, stream>>>(rowp, dinv, E);
  deg_finalize_kernel<<<(N + 255) / 256, 256, 0, stream>>>(dinv, N);

  // 2) layer 1: h1 = x @ W1 ; agg1 = scatter(norm * h1) + b1 (ReLU fused later)
  {
    int grid = nTiles < 640 ? nTiles : 640;
    gcn_gemm_kernel<128, false><<<grid, 256, 0, stream>>>(x, W1, h1, N, nTiles);
  }
  agg_init_kernel<128><<<((N * 32) + 255) / 256, 256, 0, stream>>>(h1, dinv, b1, agg1, N);
  {
    long long work = (long long)E * 32;
    agg_edge_kernel<128><<<(int)((work + 255) / 256), 256, 0, stream>>>(h1, rowp, colp, dinv, agg1, E);
  }

  // 3) layer 2: h2 = relu(agg1) @ W2 ; out = scatter(norm * h2) + b2
  {
    int grid = nTiles < 640 ? nTiles : 640;
    gcn_gemm_kernel<64, true><<<grid, 128, 0, stream>>>(agg1, W2, h2, N, nTiles);
  }
  agg_init_kernel<64><<<((N * 16) + 255) / 256, 256, 0, stream>>>(h2, dinv, b2, out, N);
  {
    long long work = (long long)E * 16;
    agg_edge_kernel<64><<<(int)((work + 255) / 256), 256, 0, stream>>>(h2, rowp, colp, dinv, out, E);
  }
}
